// ScaledAttention_70944269796054
// MI455X (gfx1250) — matrix operand order repaired and checked by reading, not compiled, not run on hardware
//
#include <hip/hip_runtime.h>

#define TT    32
#define EMB   256
#define HEAD  64
#define NWAVE 16      // 512 threads/block

// ---- WMMA fragment types (CDNA5 wave32) ----
typedef __attribute__((ext_vector_type(16))) __bf16 v16bf;
typedef __attribute__((ext_vector_type(8)))  float  v8f;

union BF16x16 {
  v16bf v;
  __bf16 h[16];
  uint4 q[2];
};
union BF16x8 {
  __bf16 h[8];
  uint4 q;
};

// native packed bf16 convert path (v_cvt_pk_bf16_f32 on gfx1250)
__device__ __forceinline__ void cvt8(__bf16* d, const float* p) {
  const float4* q = (const float4*)p;
  float4 a = q[0], b = q[1];
  d[0] = (__bf16)a.x; d[1] = (__bf16)a.y; d[2] = (__bf16)a.z; d[3] = (__bf16)a.w;
  d[4] = (__bf16)b.x; d[5] = (__bf16)b.y; d[6] = (__bf16)b.z; d[7] = (__bf16)b.w;
}

__device__ __forceinline__ v8f wmma_bf16(v16bf a, v16bf b, v8f c) {
  // D = A*B + C, fp32 accumulate
  return __builtin_amdgcn_wmma_f32_16x16x32_bf16(false, a, false, b,
                                                 (short)0, c, false, false);
}

// ---- dynamic LDS layout (bytes) ----
// W^T bf16 [3][64][256] staged once per block, then per-wave scratch with
// lifetime-based overlays:
//   bufA: Q bf16 [32][72]  -> overlaid by WEI f32 [32][33] (Q is in regs by then)
//   bufB: K bf16 [32][72]  -> overlaid by P bf16 [32][40]  (K consumed by then)
//   bufC: VT bf16 [64][40]
#define SM_WT    0
#define SM_WV0   98304
#define OFF_A    0
#define OFF_B    4608
#define OFF_C    9216
#define PW       14336
#define SMEM_TOTAL (SM_WV0 + NWAVE * PW)   // 327680 B == 320 KB (WGP max)

__global__ void __launch_bounds__(512, 1)
attn_wmma_bf16_kernel(const float* __restrict__ X,
                      const float* __restrict__ Wq, const float* __restrict__ bq,
                      const float* __restrict__ Wk, const float* __restrict__ bk,
                      const float* __restrict__ Wv, const float* __restrict__ bv,
                      float* __restrict__ Out, int nBatch)
{
  extern __shared__ char smem[];
  const int lane  = threadIdx.x & 31;
  const int wave  = threadIdx.x >> 5;
  const int hi    = lane >> 4;      // half-wave select
  const int c     = lane & 15;      // tile row/column index
  const int hi8   = hi * 8;         // A-matrix K interleave offset
  const int khalf = hi * 16;        // K half offset (unused; B uses hi8 too)

  __bf16* sWT = (__bf16*)(smem + SM_WT);

  // ---- stage W^T as bf16 into LDS, once per block ----
  {
    const float* Ws[3] = {Wq, Wk, Wv};
    for (int e = threadIdx.x; e < 3 * EMB * HEAD; e += blockDim.x) {
      int p = e >> 14;          // / (256*64)
      int r = e & 16383;
      int k = r >> 6;           // / 64
      int h = r & 63;
      sWT[p * 16384 + h * 256 + k] = (__bf16)Ws[p][r];
    }
  }
  __syncthreads();

  char* wb = smem + SM_WV0 + wave * PW;
  __bf16* sQ   = (__bf16*)(wb + OFF_A);   // [32][72]
  __bf16* sK   = (__bf16*)(wb + OFF_B);   // [32][72]
  __bf16* sVT  = (__bf16*)(wb + OFF_C);   // [64][40], V transposed
  float*  sWei = (float*) (wb + OFF_A);   // [32][33], overlays Q
  __bf16* sP   = (__bf16*)(wb + OFF_B);   // [32][40], overlays K

  const int gw     = blockIdx.x * NWAVE + wave;
  const int stride = gridDim.x * NWAVE;

  for (int b = gw; b < nBatch; b += stride) {
    const float* Xb = X + (size_t)b * (TT * EMB);
    if (b + stride < nBatch)
      __builtin_prefetch(X + (size_t)(b + stride) * (TT * EMB) + lane * 16, 0, 1);

    // ---- projections, M-tile outermost to halve X-fragment registers ----
    #pragma unroll
    for (int m = 0; m < 2; ++m) {
      // X A-fragments for this M-tile (bf16), X read once per (m)
      BF16x16 XA[8];
      const float* xr = Xb + (16 * m + c) * EMB;
      #pragma unroll
      for (int j = 0; j < 8; ++j) {
        cvt8(&XA[j].h[0], xr + 32 * j + hi8);        // K = 32j+hi8 .. +7
        cvt8(&XA[j].h[8], xr + 32 * j + 16 + hi8);   // K = 32j+16+hi8 .. +7
      }

      #pragma unroll
      for (int p = 0; p < 3; ++p) {
        const float* bias = (p == 0) ? bq : (p == 1) ? bk : bv;
        const __bf16* wt = sWT + p * 16384;
        #pragma unroll
        for (int n = 0; n < 4; ++n) {
          int col = 16 * n + c;
          float bc = bias[col];
          v8f acc;
          #pragma unroll
          for (int v = 0; v < 8; ++v) acc[v] = bc;   // bias folded into C init
          const __bf16* wrow = wt + col * 256 + hi8;
          #pragma unroll
          for (int j = 0; j < 8; ++j) {
            BF16x16 bb;
            bb.q[0] = *(const uint4*)(wrow + 32 * j);
            bb.q[1] = *(const uint4*)(wrow + 32 * j + 16);
            acc = wmma_bf16(XA[j].v, bb.v, acc);
          }
          if (p == 0) {
            #pragma unroll
            for (int v = 0; v < 8; ++v)
              sQ[(v + 8 * hi + 16 * m) * 72 + col] = (__bf16)acc[v];
          } else if (p == 1) {
            #pragma unroll
            for (int v = 0; v < 8; ++v)
              sK[(v + 8 * hi + 16 * m) * 72 + col] = (__bf16)acc[v];
          } else {
            // V transposed: rows along v are consecutive -> single b128 store
            BF16x8 t;
            #pragma unroll
            for (int v = 0; v < 8; ++v) t.h[v] = (__bf16)acc[v];
            *(uint4*)(sVT + col * 40 + 8 * hi + 16 * m) = t.q;
          }
        }
      }
    }

    // ---- wei = Q K^T  (M=32, N=32, K=64) ----
    BF16x16 QA[2][2];
    #pragma unroll
    for (int m = 0; m < 2; ++m) {
      const __bf16* qr = sQ + (16 * m + c) * 72;
      #pragma unroll
      for (int j = 0; j < 2; ++j) {
        QA[m][j].q[0] = *(const uint4*)(qr + 32 * j + hi8);
        QA[m][j].q[1] = *(const uint4*)(qr + 32 * j + 16 + hi8);
      }
    }
    #pragma unroll
    for (int ns = 0; ns < 2; ++ns) {
      BF16x16 KB[2];
      const __bf16* kr = sK + (16 * ns + c) * 72 + hi8;
      KB[0].q[0] = *(const uint4*)(kr);
      KB[0].q[1] = *(const uint4*)(kr + 16);
      KB[1].q[0] = *(const uint4*)(kr + 32);
      KB[1].q[1] = *(const uint4*)(kr + 48);
      #pragma unroll
      for (int mt = 0; mt < 2; ++mt) {
        v8f w = {};
        w = wmma_bf16(QA[mt][0].v, KB[0].v, w);
        w = wmma_bf16(QA[mt][1].v, KB[1].v, w);
        #pragma unroll
        for (int v = 0; v < 8; ++v)
          sWei[(v + 8 * hi + 16 * mt) * 33 + 16 * ns + c] = w[v];  // overlays Q
      }
    }

    // ---- causal softmax (mask-then-scale, per reference); lane t owns row t ----
    {
      const int t = lane;
      const float scale = 0.125f;   // 1/sqrt(64)
      float mx = -3.0e38f;
      #pragma unroll
      for (int s = 0; s < TT; ++s)
        if (s <= t) mx = fmaxf(mx, sWei[t * 33 + s] * scale);
      float sum = 0.f;
      #pragma unroll
      for (int s = 0; s < TT; ++s) {
        float e = (s <= t) ? __expf(sWei[t * 33 + s] * scale - mx) : 0.f;
        sWei[t * 33 + s] = e;
        sum += e;
      }
      float inv = 1.0f / sum;
      #pragma unroll
      for (int s = 0; s < TT; ++s)
        sP[t * 40 + s] = (__bf16)(sWei[t * 33 + s] * inv);         // overlays K
    }

    // ---- out = P V  (M=32, N=64, K=32) ----
    BF16x16 PA[2];
    #pragma unroll
    for (int m = 0; m < 2; ++m) {
      const __bf16* pr = sP + (16 * m + c) * 40;
      PA[m].q[0] = *(const uint4*)(pr + hi8);
      PA[m].q[1] = *(const uint4*)(pr + 16 + hi8);
    }
    float* Ob = Out + (size_t)b * (TT * HEAD);
    #pragma unroll
    for (int n = 0; n < 4; ++n) {
      BF16x16 vb;
      const __bf16* vr = sVT + (16 * n + c) * 40 + hi8;
      vb.q[0] = *(const uint4*)(vr);
      vb.q[1] = *(const uint4*)(vr + 16);
      v8f o0 = {}; v8f o1 = {};
      o0 = wmma_bf16(PA[0].v, vb.v, o0);
      o1 = wmma_bf16(PA[1].v, vb.v, o1);
      #pragma unroll
      for (int v = 0; v < 8; ++v) {
        Ob[(v + 8 * hi)      * HEAD + 16 * n + c] = o0[v];
        Ob[(v + 8 * hi + 16) * HEAD + 16 * n + c] = o1[v];
      }
    }
  }
}

extern "C" void kernel_launch(void* const* d_in, const int* in_sizes, int n_in,
                              void* d_out, int out_size, void* d_ws, size_t ws_size,
                              hipStream_t stream) {
  const float* X  = (const float*)d_in[0];
  const float* Wq = (const float*)d_in[1];
  const float* bq = (const float*)d_in[2];
  const float* Wk = (const float*)d_in[3];
  const float* bk = (const float*)d_in[4];
  const float* Wv = (const float*)d_in[5];
  const float* bv = (const float*)d_in[6];
  float* Out = (float*)d_out;

  int nBatch = in_sizes[0] / (TT * EMB);   // 8192
  int blocks = 512;                        // 512 blocks x 16 waves -> 1 batch/wave
  attn_wmma_bf16_kernel<<<dim3(blocks), dim3(512), SMEM_TOTAL, stream>>>(
      X, Wq, bq, Wk, bk, Wv, bv, Out, nBatch);
}
